// GMLLMSelfAttention_15367392985632
// MI455X (gfx1250) — matrix operand order repaired and checked by reading, not compiled, not run on hardware
//
#include <hip/hip_runtime.h>
#include <hip/hip_fp16.h>

// Problem constants (from reference)
#define B_  4
#define S_  2048
#define D_  768
#define H_  12
#define HD_ 64

typedef __attribute__((ext_vector_type(16))) _Float16 v16h;
typedef __attribute__((ext_vector_type(8)))  _Float16 v8h;
typedef __attribute__((ext_vector_type(8)))  float    v8f;
typedef __attribute__((ext_vector_type(4)))  int      v4i;

// ---- gfx1250 async global->LDS path (guarded; fallback = regular ld/st) ----
// Probe-verified signature: (v4i* global_src, v4i* lds_dst, imm offset, imm cpol)
#if defined(__gfx1250__) && __has_builtin(__builtin_amdgcn_global_load_async_to_lds_b128)
#define USE_ASYNC_LDS 1
#define ASYNC_CP16(gp, lp) \
  __builtin_amdgcn_global_load_async_to_lds_b128((v4i*)(gp), (v4i*)(lp), 0, 0)
#if __has_builtin(__builtin_amdgcn_s_wait_asynccnt)
#define ASYNC_WAIT() __builtin_amdgcn_s_wait_asynccnt(0)
#else
#define ASYNC_WAIT() asm volatile("s_wait_asynccnt 0x0" ::: "memory")
#endif
#else
#define USE_ASYNC_LDS 0
#define ASYNC_WAIT() ((void)0)
#endif

static __device__ __forceinline__ v16h pack2(v8h lo, v8h hi) {
  v16h r;
#pragma unroll
  for (int i = 0; i < 8; ++i) { r[i] = lo[i]; r[i + 8] = hi[i]; }
  return r;
}

static __device__ __forceinline__ v8f vzero8() {
  v8f z;
#pragma unroll
  for (int i = 0; i < 8; ++i) z[i] = 0.0f;
  return z;
}

static __device__ __forceinline__ v8f wmma_f16(v16h a, v16h b, v8f c) {
  return __builtin_amdgcn_wmma_f32_16x16x32_f16(false, a, false, b, (short)0, c,
                                                false, false);
}

// ---------------------------------------------------------------------------
// Kernel 0a: plain f32 -> f16 convert (hidden_states, weights)
// ---------------------------------------------------------------------------
__global__ void cvt_f16(const float* __restrict__ src,
                        _Float16* __restrict__ dst, int n) {
  int i = blockIdx.x * blockDim.x + threadIdx.x;
  if (i < n) dst[i] = (_Float16)src[i];
}

// ---------------------------------------------------------------------------
// Kernel 0b: layout streams f32 [B,S,D] -> f16 [B,H,S,HD]
// ---------------------------------------------------------------------------
__global__ void cvt_split_heads(const float* __restrict__ src,
                                _Float16* __restrict__ dst) {
  int i = blockIdx.x * blockDim.x + threadIdx.x;  // flat over B*H*S*HD
  int hd = i & (HD_ - 1);
  int t  = i >> 6;
  int s  = t & (S_ - 1);
  int t2 = t >> 11;
  int h  = t2 % H_;
  int b  = t2 / H_;
  dst[i] = (_Float16)src[((size_t)(b * S_ + s)) * D_ + h * HD_ + hd];
}

// ---------------------------------------------------------------------------
// Kernel 1: QKV projection  out[m,n] = sum_d hs[m,d]*W[n,d] + bias[n]
// Inputs already f16 (hws: [B*S, D], wws: 3x[D, D]).
// Tile: 128(M) x 64(N), K-step 32, f16 WMMA, f32 accumulate.
// Double-buffered async DMA pipeline: one barrier / k-step, DMA of tile i+1
// overlaps WMMA of tile i.  Output f16 [B,H,S,HD]; blockIdx.z selects q/k/v.
// ---------------------------------------------------------------------------
__global__ __launch_bounds__(256, 2) void qkv_proj(
    const _Float16* __restrict__ hws, const _Float16* __restrict__ wws,
    const float* __restrict__ bq, const float* __restrict__ bk,
    const float* __restrict__ bv, _Float16* __restrict__ qws,
    _Float16* __restrict__ kws, _Float16* __restrict__ vws) {
  const int z = blockIdx.z;
  const _Float16* W = wws + (size_t)z * D_ * D_;
  const float* bias = (z == 0) ? bq : (z == 1) ? bk : bv;
  _Float16* dst     = (z == 0) ? qws : (z == 1) ? kws : vws;

  const int m0 = blockIdx.x * 128;
  const int n0 = blockIdx.y * 64;
  const int t  = threadIdx.x;
  const int wave = t >> 5;
  const int lane = t & 31;
  const int ln = lane & 15;
  const int hi = lane >> 4;

  // stride 48 halves (96B): every 8-half chunk stays 16B aligned for b128 ops
  __shared__ _Float16 sa[2][128 * 48];
  __shared__ _Float16 sb[2][64 * 48];

  // stage A (128x32 halves, 512 b128 chunks) + B (64x32, 256 chunks)
  auto stage = [&](int buf, int kk) {
#pragma unroll
    for (int i = 0; i < 2; ++i) {
      int f = t + i * 256;           // 0..511
      int row = f >> 2, c8 = f & 3;  // 4 chunks of 8 halves per 32-wide row
#if USE_ASYNC_LDS
      ASYNC_CP16(&hws[(size_t)(m0 + row) * D_ + kk + c8 * 8],
                 &sa[buf][row * 48 + c8 * 8]);
#else
      *(v8h*)&sa[buf][row * 48 + c8 * 8] =
          *(const v8h*)&hws[(size_t)(m0 + row) * D_ + kk + c8 * 8];
#endif
    }
    {
      int row = t >> 2, c8 = t & 3;
#if USE_ASYNC_LDS
      ASYNC_CP16(&W[(size_t)(n0 + row) * D_ + kk + c8 * 8],
                 &sb[buf][row * 48 + c8 * 8]);
#else
      *(v8h*)&sb[buf][row * 48 + c8 * 8] =
          *(const v8h*)&W[(size_t)(n0 + row) * D_ + kk + c8 * 8];
#endif
    }
  };

  v8f acc[4];
#pragma unroll
  for (int i = 0; i < 4; ++i) acc[i] = vzero8();

  const int nk = D_ / 32;  // 24 k-steps
  stage(0, 0);
  for (int it = 0; it < nk; ++it) {
    ASYNC_WAIT();        // my chunk-it transfers have landed
    __syncthreads();     // everyone's landed; everyone done reading buf_next
    if (it + 1 < nk) stage((it + 1) & 1, (it + 1) * 32);  // overlaps compute

    const _Float16* A  = sa[it & 1];
    const _Float16* Bt = sb[it & 1];
    // A frag (16x32): lane ln = row, K halves {hi*8..+7, hi*8+16..+23}
    const int mrow = wave * 16 + ln;
    const int kba  = hi * 8;
    v16h a = pack2(*(const v8h*)&A[mrow * 48 + kba],
                   *(const v8h*)&A[mrow * 48 + kba + 16]);
#pragma unroll
    for (int nt = 0; nt < 4; ++nt) {
      // B frag (32x16): lane = column n, K halves {hi*8..+7, hi*8+16..+23}
      const int ncol = nt * 16 + ln;
      const int kbb  = hi * 8;
      v16h bfr = pack2(*(const v8h*)&Bt[ncol * 48 + kbb],
                       *(const v8h*)&Bt[ncol * 48 + kbb + 16]);
      acc[nt] = wmma_f16(a, bfr, acc[nt]);
    }
  }

  // epilogue: C layout -> [B,H,S,HD] f16 with bias
#pragma unroll
  for (int nt = 0; nt < 4; ++nt) {
    int n_g = n0 + nt * 16 + ln;
    int hh = n_g >> 6, hd = n_g & 63;
    float bb_ = bias[n_g];
#pragma unroll
    for (int r = 0; r < 8; ++r) {
      int m_g = m0 + wave * 16 + hi * 8 + r;
      int bb = m_g >> 11, ss = m_g & (S_ - 1);
      dst[((size_t)(bb * H_ + hh) * S_ + ss) * HD_ + hd] =
          (_Float16)(acc[nt][r] + bb_);
    }
  }
}

// ---------------------------------------------------------------------------
// Kernel 2: fused dual-stream flash attention.
// Block = 256 threads (8 waves), one (b,h), 128 query rows (16 per wave).
// Single pass over keys in chunks of 32 with online softmax; double-buffered
// async staging of K/LK (DMA) + V (store-side transpose), one barrier/chunk.
// scores = (q.kT + lq.lkT)/8 + mask ; ctx = softmax(scores) @ v
// ---------------------------------------------------------------------------
__global__ __launch_bounds__(256, 2) void flash_attn(
    const _Float16* __restrict__ qws, const _Float16* __restrict__ kws,
    const _Float16* __restrict__ vws, const _Float16* __restrict__ lqws,
    const _Float16* __restrict__ lkws, const float* __restrict__ mask,
    float* __restrict__ out) {
  const int q0 = blockIdx.x * 128;
  const int hh = blockIdx.y;
  const int bb = blockIdx.z;
  const int t = threadIdx.x;
  const int wave = t >> 5;
  const int lane = t & 31;
  const int ln = lane & 15;
  const int hi = lane >> 4;
  const size_t head_base = (size_t)(bb * H_ + hh) * S_ * HD_;

  __shared__ _Float16 sk [2][32 * 72];   // K chunk, row-major (stride 144B)
  __shared__ _Float16 slk[2][32 * 72];   // layout-K chunk
  __shared__ _Float16 svt[2][64 * 48];   // V chunk transposed [d][key]
  __shared__ _Float16 sp [8 * 16 * 48];  // per-wave P (probs) staging

  auto stage = [&](int buf, int j0) {
    int key = t >> 3, c8 = t & 7;  // 32 keys x 8 chunks of 8 halves
    size_t g = head_base + (size_t)(j0 + key) * HD_ + c8 * 8;
#if USE_ASYNC_LDS
    ASYNC_CP16(&kws[g],  &sk[buf][key * 72 + c8 * 8]);
    ASYNC_CP16(&lkws[g], &slk[buf][key * 72 + c8 * 8]);
#else
    *(v8h*)&sk[buf][key * 72 + c8 * 8]  = *(const v8h*)&kws[g];
    *(v8h*)&slk[buf][key * 72 + c8 * 8] = *(const v8h*)&lkws[g];
#endif
    v8h v = *(const v8h*)&vws[g];
#pragma unroll
    for (int j = 0; j < 8; ++j) svt[buf][(c8 * 8 + j) * 48 + key] = v[j];
    __builtin_prefetch(&vws[g + 32 * HD_], 0, 1);  // global_prefetch_b8 hint
  };

  // Q / layout-Q fragments (A layout), HD=64 -> two K-steps of 32
  v16h aq[2], alq[2];
  {
    const size_t qrow = head_base + (size_t)(q0 + wave * 16 + ln) * HD_;
#pragma unroll
    for (int ks = 0; ks < 2; ++ks) {
      int kb = hi * 8 + ks * 32;
      aq[ks]  = pack2(*(const v8h*)&qws[qrow + kb],
                      *(const v8h*)&qws[qrow + kb + 16]);
      alq[ks] = pack2(*(const v8h*)&lqws[qrow + kb],
                      *(const v8h*)&lqws[qrow + kb + 16]);
    }
  }

  v8f acc[4];
#pragma unroll
  for (int i = 0; i < 4; ++i) acc[i] = vzero8();
  float m_row[8], l_row[8];
#pragma unroll
  for (int r = 0; r < 8; ++r) { m_row[r] = -INFINITY; l_row[r] = 0.0f; }
  const float scale = 0.125f;  // 1/sqrt(64)

  const int nchunks = S_ / 32;  // 64
  stage(0, 0);
  for (int jc = 0; jc < nchunks; ++jc) {
    ASYNC_WAIT();     // my chunk-jc DMA landed
    __syncthreads();  // everyone's landed; prior-iteration reads all retired
    if (jc + 1 < nchunks) stage((jc + 1) & 1, (jc + 1) * 32);  // overlap

    const int buf = jc & 1;
    const int j0  = jc * 32;

    // ---- dual-stream scores: two 16x16 C tiles (32 keys) ----
    v8f sfrag[2];
#pragma unroll
    for (int nh = 0; nh < 2; ++nh) {
      v8f c = vzero8();
#pragma unroll
      for (int ks = 0; ks < 2; ++ks) {
        int krow = nh * 16 + ln;     // lane = key column of B
        int kb = hi * 8 + ks * 32;  // K slice of HD
        v16h bk = pack2(*(const v8h*)&sk[buf][krow * 72 + kb],
                        *(const v8h*)&sk[buf][krow * 72 + kb + 16]);
        c = wmma_f16(aq[ks], bk, c);
        v16h bl = pack2(*(const v8h*)&slk[buf][krow * 72 + kb],
                        *(const v8h*)&slk[buf][krow * 72 + kb + 16]);
        c = wmma_f16(alq[ks], bl, c);
      }
      float mv = mask[(size_t)bb * S_ + j0 + nh * 16 + ln];
#pragma unroll
      for (int r = 0; r < 8; ++r) c[r] = c[r] * scale + mv;
      sfrag[nh] = c;
    }

    // ---- online softmax (rows in VGPR index; reduce across 16-lane half) --
    float mx[8];
#pragma unroll
    for (int r = 0; r < 8; ++r) mx[r] = fmaxf(sfrag[0][r], sfrag[1][r]);
#pragma unroll
    for (int off = 1; off < 16; off <<= 1)
#pragma unroll
      for (int r = 0; r < 8; ++r)
        mx[r] = fmaxf(mx[r], __shfl_xor(mx[r], off, 32));

    float alpha[8];
#pragma unroll
    for (int r = 0; r < 8; ++r) {
      float mn = fmaxf(m_row[r], mx[r]);
      alpha[r] = __expf(m_row[r] - mn);
      m_row[r] = mn;
    }
#pragma unroll
    for (int nh = 0; nh < 2; ++nh)
#pragma unroll
      for (int r = 0; r < 8; ++r)
        sfrag[nh][r] = __expf(sfrag[nh][r] - m_row[r]);

    float rs[8];
#pragma unroll
    for (int r = 0; r < 8; ++r) rs[r] = sfrag[0][r] + sfrag[1][r];
#pragma unroll
    for (int off = 1; off < 16; off <<= 1)
#pragma unroll
      for (int r = 0; r < 8; ++r) rs[r] += __shfl_xor(rs[r], off, 32);
#pragma unroll
    for (int r = 0; r < 8; ++r) l_row[r] = l_row[r] * alpha[r] + rs[r];
#pragma unroll
    for (int i = 0; i < 4; ++i)
#pragma unroll
      for (int r = 0; r < 8; ++r) acc[i][r] *= alpha[r];

    // ---- P: C layout -> A layout via per-wave LDS round trip ----
    _Float16* pw = &sp[wave * 16 * 48];
#pragma unroll
    for (int nh = 0; nh < 2; ++nh)
#pragma unroll
      for (int r = 0; r < 8; ++r)
        pw[(hi * 8 + r) * 48 + nh * 16 + ln] = (_Float16)sfrag[nh][r];

    const int kbp = hi * 8;
    v16h aP = pack2(*(const v8h*)&pw[ln * 48 + kbp],
                    *(const v8h*)&pw[ln * 48 + kbp + 16]);

    // ---- ctx += P(16x32) @ V(32x64) ----
#pragma unroll
    for (int nv = 0; nv < 4; ++nv) {
      int d = nv * 16 + ln;  // lane = output feature column
      int kb2 = hi * 8;     // key slice
      v16h bV = pack2(*(const v8h*)&svt[buf][d * 48 + kb2],
                      *(const v8h*)&svt[buf][d * 48 + kb2 + 16]);
      acc[nv] = wmma_f16(aP, bV, acc[nv]);
    }
  }

  // ---- epilogue: normalize and write [B,S,D] f32 ----
#pragma unroll
  for (int nv = 0; nv < 4; ++nv) {
    int d = nv * 16 + ln;
#pragma unroll
    for (int r = 0; r < 8; ++r) {
      int sq = q0 + wave * 16 + hi * 8 + r;
      out[((size_t)(bb * S_) + sq) * D_ + hh * HD_ + d] = acc[nv][r] / l_row[r];
    }
  }
}

// ---------------------------------------------------------------------------
extern "C" void kernel_launch(void* const* d_in, const int* in_sizes, int n_in,
                              void* d_out, int out_size, void* d_ws,
                              size_t ws_size, hipStream_t stream) {
  const float* hs   = (const float*)d_in[0];
  const float* lq   = (const float*)d_in[1];
  const float* lk   = (const float*)d_in[2];
  const float* mask = (const float*)d_in[3];
  const float* Wq   = (const float*)d_in[4];
  const float* bq   = (const float*)d_in[5];
  const float* Wk   = (const float*)d_in[6];
  const float* bk   = (const float*)d_in[7];
  const float* Wv   = (const float*)d_in[8];
  const float* bv   = (const float*)d_in[9];
  float* out = (float*)d_out;

  const size_t per = (size_t)B_ * H_ * S_ * HD_;  // == B*S*D elements
  const size_t wsz = (size_t)D_ * D_;             // one weight matrix
  _Float16* ws   = (_Float16*)d_ws;
  _Float16* qws  = ws;
  _Float16* kws  = ws + per;
  _Float16* vws  = ws + 2 * per;
  _Float16* lqws = ws + 3 * per;
  _Float16* lkws = ws + 4 * per;
  _Float16* hws  = ws + 5 * per;        // hidden_states f16 [B*S, D]
  _Float16* wws  = ws + 6 * per;        // Wq|Wk|Wv f16, each [D, D]

  dim3 blk(256);
  // one-shot f32->f16 conversions (bandwidth-trivial)
  cvt_f16<<<dim3((int)(per / 256)), blk, 0, stream>>>(hs, hws, (int)per);
  cvt_f16<<<dim3((int)(wsz / 256)), blk, 0, stream>>>(Wq, wws, (int)wsz);
  cvt_f16<<<dim3((int)(wsz / 256)), blk, 0, stream>>>(Wk, wws + wsz, (int)wsz);
  cvt_f16<<<dim3((int)(wsz / 256)), blk, 0, stream>>>(Wv, wws + 2 * wsz,
                                                      (int)wsz);
  cvt_split_heads<<<dim3((int)(per / 256)), blk, 0, stream>>>(lq, lqws);
  cvt_split_heads<<<dim3((int)(per / 256)), blk, 0, stream>>>(lk, lkws);

  // QKV projections via WMMA
  {
    dim3 grid((B_ * S_) / 128, D_ / 64, 3);
    qkv_proj<<<grid, blk, 0, stream>>>(hws, wws, bq, bk, bv, qws, kws, vws);
  }
  // fused dual-stream flash attention
  {
    dim3 grid(S_ / 128, H_, B_);
    flash_attn<<<grid, blk, 0, stream>>>(qws, kws, vws, lqws, lkws, mask, out);
  }
}
